// GGNNModelClassification_33139967656251
// MI455X (gfx1250) — matrix operand order repaired and checked by reading, not compiled, not run on hardware
//
#include <hip/hip_runtime.h>
#include <hip/hip_bf16.h>

// GGNN on MI455X (gfx1250): bf16 WMMA 16x16x32, fp32 accumulation.
// All matrix operands live in LDS/global as bf16 so GEMM fragments are pure
// ds_load_b128 / global_load_b128 (no fp32->bf16 converts in the inner loop).
// 1 block per (s,b); 8 waves; wave w owns output rows [16w,16w+16).
// Gate nonlinearities use native v_exp/v_rcp (co-executes with WMMA).
// Workspace (bf16): adjacency [B,E,N,N] + two h buffers [S,B,N,U]  (~34 MB).

#define Bv 32
#define Sv 32
#define Nn 128
#define Uu 64
#define Ee 2
#define NCLS 5

#define HLD 72    // padded row stride for h / a / rh (bf16): 144B = 36 dw -> no bank conflicts
#define WLD 72    // padded row stride for transposed weight slots
#define MLD 136   // padded row stride for m^T [u][node]: 272B = 68 dw -> no bank conflicts
#define SCR (Nn * HLD)   // 9216 bf16: covers both m^T (64*136=8704) and a/rh (128*72)

typedef __attribute__((ext_vector_type(8)))  float  v8f;
typedef __attribute__((ext_vector_type(8)))  __bf16 v8bf;
typedef __attribute__((ext_vector_type(16))) __bf16 v16bf;

__device__ __forceinline__ v8f vzero8() {
    v8f z;
#pragma unroll
    for (int i = 0; i < 8; ++i) z[i] = 0.0f;
    return z;
}

// ---- fragment builders per CDNA5 ISA 7.12.2 (wave32) ---------------------
// A 16x32 (16-bit): lane<16 -> M=lane, K = {k0..k0+7, k0+16..k0+23};
// lane>=16 -> M=lane-16, K groups shifted by +8.  Two 16B contiguous runs.
__device__ __forceinline__ v16bf afrag(const __bf16* __restrict__ src, int ld,
                                       int row0, int k0, int lane) {
    const int m    = row0 + (lane & 15);
    const int koff = (lane >> 4) << 3;                 // 0 or 8
    const __bf16* p = src + m * ld + k0 + koff;
    const v8bf lo = *(const v8bf*)p;                   // K = base .. base+7
    const v8bf hi = *(const v8bf*)(p + 16);            // K = base+16 .. base+23
    v16bf a;
#pragma unroll
    for (int t = 0; t < 8; ++t) { a[t] = lo[t]; a[t + 8] = hi[t]; }
    return a;
}

// B 32x16 (16-bit) from a TRANSPOSED source BT[n][k] (ld = ldt):
// lane<16 -> N=n0+lane, K={k0..k0+7, k0+16..k0+23}; lane>=16 -> N=n0+lane-16, K groups shifted by +8.
__device__ __forceinline__ v16bf bfragT(const __bf16* __restrict__ srcT, int ldt,
                                        int k0, int n0, int lane) {
    const __bf16* p = srcT + (size_t)(n0 + (lane & 15)) * ldt + k0 + ((lane >> 4) << 3);
    const v8bf lo = *(const v8bf*)p;
    const v8bf hi = *(const v8bf*)(p + 16);
    v16bf b;
#pragma unroll
    for (int t = 0; t < 8; ++t) { b[t] = lo[t]; b[t + 8] = hi[t]; }
    return b;
}

// acc[nt] (N = 0..63) += A[row0:row0+16, 0:K] @ B[0:K, 0:64] (B given transposed)
__device__ __forceinline__ void gemm_bb(v8f acc[4], const __bf16* __restrict__ A, int lda,
                                        int row0, const __bf16* __restrict__ BT, int ldt,
                                        int K, int lane) {
    for (int k0 = 0; k0 < K; k0 += 32) {
        const v16bf af = afrag(A, lda, row0, k0, lane);
#pragma unroll
        for (int nt = 0; nt < 4; ++nt) {
            const v16bf bf = bfragT(BT, ldt, k0, nt * 16, lane);
            acc[nt] = __builtin_amdgcn_wmma_f32_16x16x32_bf16(
                false, af, false, bf, (short)0, acc[nt], false, false);
        }
    }
}

// stage one [U][U] fp32 weight matrix into an LDS slot, transposed, bf16
__device__ __forceinline__ void stageT(__bf16* dst, const float* __restrict__ src, int tid) {
    for (int i = tid; i < Uu * Uu; i += 256) {
        const int r = i >> 6;       // k (row of W)
        const int c = i & 63;       // n (col of W)
        dst[c * WLD + r] = (__bf16)src[i];
    }
}

// native-transcendental activations: v_exp_f32 + v_rcp_f32, co-execute with WMMA
__device__ __forceinline__ float fast_sigmoid(float x) {
    return __builtin_amdgcn_rcpf(1.0f + __expf(-x));
}
__device__ __forceinline__ float fast_tanh(float x) {
    // tanh(x) = 1 - 2/(exp(2x)+1); saturates to +/-1 for large |x|
    return 1.0f - 2.0f * __builtin_amdgcn_rcpf(1.0f + __expf(2.0f * x));
}

// --------------------------------------------------------------------------
__global__ __launch_bounds__(256) void ggnn_step(
    const __bf16* __restrict__ h_in, __bf16* __restrict__ h_out,
    const __bf16* __restrict__ adjb,                     // [B,E,N,N] bf16
    const float* __restrict__ Wmsg,                      // [E,U,U]
    const float* __restrict__ bmsg,                      // [U]
    const float* __restrict__ Wg,                        // [3,U,U]
    const float* __restrict__ Ug,                        // [3,U,U]
    const float* __restrict__ bgv)                       // [3,U]
{
    __shared__ __bf16 sh_h[Nn * HLD];        // 18 KB
    __shared__ __bf16 sh_w[2 * Uu * WLD];    // 18 KB: two transposed weight slots
    __shared__ __bf16 sh_scr[SCR];           // 18 KB: m^T -> a -> r*h

    const int tid  = threadIdx.x;
    const int lane = tid & 31;
    const int wave = tid >> 5;
    const int row0 = wave * 16;
    const int p    = blockIdx.x;             // s*B + b
    const int b    = p & (Bv - 1);

    const __bf16* hsrc = h_in  + (size_t)p * Nn * Uu;
    __bf16*       hdst = h_out + (size_t)p * Nn * Uu;

    // load h (dword copies into padded rows) + stage message weights
    {
        const unsigned int* src = (const unsigned int*)hsrc;
        for (int i = tid; i < Nn * Uu / 2; i += 256) {
            const int r = i >> 5, c2 = (i & 31) << 1;
            *(unsigned int*)&sh_h[r * HLD + c2] = src[i];
        }
    }
    stageT(sh_w, Wmsg, tid);
    stageT(sh_w + Uu * WLD, Wmsg + Uu * Uu, tid);
    __syncthreads();

    const int nloc = lane & 15;
    const int mhi  = (lane >> 4) << 3;

    // ---- t = sum_e A_e @ (h @ W_e) ---------------------------------------
    v8f acc[4];
#pragma unroll
    for (int nt = 0; nt < 4; ++nt) acc[nt] = vzero8();

    __bf16* sh_mT = sh_scr;                  // [u][node], ld = MLD
    for (int e = 0; e < Ee; ++e) {
        v8f macc[4];
#pragma unroll
        for (int nt = 0; nt < 4; ++nt) macc[nt] = vzero8();
        gemm_bb(macc, sh_h, HLD, row0, sh_w + e * Uu * WLD, WLD, Uu, lane);

        __syncthreads();                     // prev m^T readers done
#pragma unroll
        for (int nt = 0; nt < 4; ++nt) {
            const int u = nt * 16 + nloc;
            v8bf mv;
#pragma unroll
            for (int v = 0; v < 8; ++v) mv[v] = (__bf16)macc[nt][v];
            *(v8bf*)(sh_mT + u * MLD + row0 + mhi) = mv;   // rows v..v+7, 16B store
        }
        __syncthreads();                     // m^T fully written

        gemm_bb(acc, adjb + ((size_t)b * Ee + e) * Nn * Nn, Nn, row0, sh_mT, MLD, Nn, lane);
    }
    __syncthreads();                         // all m^T readers done before scratch reuse

    // ---- a = t + b_msg -> LDS bf16 (own rows); restage Wg0/Ug0 -----------
    __bf16* sh_a = sh_scr;                   // row-major, ld = HLD
#pragma unroll
    for (int nt = 0; nt < 4; ++nt) {
        const int u = nt * 16 + nloc;
        const float bm = bmsg[u];
#pragma unroll
        for (int v = 0; v < 8; ++v)
            sh_a[(row0 + v + mhi) * HLD + u] = (__bf16)(acc[nt][v] + bm);
    }
    stageT(sh_w, Wg, tid);
    stageT(sh_w + Uu * WLD, Ug, tid);
    __syncthreads();

    v8f zacc[4];
#pragma unroll
    for (int nt = 0; nt < 4; ++nt) zacc[nt] = vzero8();
    gemm_bb(zacc, sh_a, HLD, row0, sh_w, WLD, Uu, lane);
    gemm_bb(zacc, sh_h, HLD, row0, sh_w + Uu * WLD, WLD, Uu, lane);

    __syncthreads();
    stageT(sh_w, Wg + 1 * Uu * Uu, tid);
    stageT(sh_w + Uu * WLD, Ug + 1 * Uu * Uu, tid);
    __syncthreads();

    v8f racc[4];
#pragma unroll
    for (int nt = 0; nt < 4; ++nt) racc[nt] = vzero8();
    gemm_bb(racc, sh_a, HLD, row0, sh_w, WLD, Uu, lane);
    gemm_bb(racc, sh_h, HLD, row0, sh_w + Uu * WLD, WLD, Uu, lane);

    __syncthreads();
    stageT(sh_w, Wg + 2 * Uu * Uu, tid);
    stageT(sh_w + Uu * WLD, Ug + 2 * Uu * Uu, tid);
    __syncthreads();

    v8f cacc[4];
#pragma unroll
    for (int nt = 0; nt < 4; ++nt) cacc[nt] = vzero8();
    gemm_bb(cacc, sh_a, HLD, row0, sh_w, WLD, Uu, lane);   // a @ Wg2

    // ---- z, r; r*h into scratch (own rows, same-wave LDS ordering) -------
    __bf16* sh_rh = sh_scr;
#pragma unroll
    for (int nt = 0; nt < 4; ++nt) {
        const int u = nt * 16 + nloc;
        const float bz = bgv[0 * Uu + u];
        const float br = bgv[1 * Uu + u];
#pragma unroll
        for (int v = 0; v < 8; ++v) {
            const int r_ = row0 + v + mhi;
            const float hval = (float)sh_h[r_ * HLD + u];
            const float z = fast_sigmoid(zacc[nt][v] + bz);
            const float r = fast_sigmoid(racc[nt][v] + br);
            zacc[nt][v] = z;
            sh_rh[r_ * HLD + u] = (__bf16)(r * hval);
        }
    }

    gemm_bb(cacc, sh_rh, HLD, row0, sh_w + Uu * WLD, WLD, Uu, lane);   // + (r*h) @ Ug2

    // ---- h' = (1-z)h + z*tanh(c) -> global bf16 --------------------------
#pragma unroll
    for (int nt = 0; nt < 4; ++nt) {
        const int u = nt * 16 + nloc;
        const float bc = bgv[2 * Uu + u];
#pragma unroll
        for (int v = 0; v < 8; ++v) {
            const int r_ = row0 + v + mhi;
            const float hval = (float)sh_h[r_ * HLD + u];
            const float c = fast_tanh(cacc[nt][v] + bc);
            const float z = zacc[nt][v];
            hdst[r_ * Uu + u] = (__bf16)((1.0f - z) * hval + z * c);
        }
    }
}

// --------------------------------------------------------------------------
// adjacency fp32 -> bf16 (same layout)
__global__ __launch_bounds__(256) void cvt_adj(const float* __restrict__ in,
                                               __bf16* __restrict__ out) {
    const size_t i = (size_t)blockIdx.x * blockDim.x + threadIdx.x;
    if (i < (size_t)Bv * Ee * Nn * Nn) out[i] = (__bf16)in[i];
}

// input_seq fp32 [B,S,N,U] -> h0 bf16 [S,B,N,U]
__global__ __launch_bounds__(256) void ggnn_init(const float* __restrict__ in,
                                                 __bf16* __restrict__ h) {
    const size_t i = (size_t)blockIdx.x * blockDim.x + threadIdx.x;
    const size_t total = (size_t)Sv * Bv * Nn * Uu;
    if (i >= total) return;
    const int u = (int)(i % Uu);
    size_t t = i / Uu;
    const int n = (int)(t % Nn); t /= Nn;
    const int b = (int)(t % Bv); t /= Bv;
    const int s = (int)t;
    h[i] = (__bf16)in[(((size_t)b * Sv + s) * Nn + n) * Uu + u];
}

// --------------------------------------------------------------------------
// last-timestep select + relu + fc + max over nodes.  1 block per batch b.
__global__ __launch_bounds__(128) void ggnn_head(const __bf16* __restrict__ hfin,
                                                 const int* __restrict__ seq_len,
                                                 const float* __restrict__ fcw,  // [U,5]
                                                 const float* __restrict__ fcb,  // [5]
                                                 float* __restrict__ out) {      // [B,5]
    __shared__ float sm[NCLS * Nn];
    const int b = blockIdx.x;
    const int n = threadIdx.x;
    int idx = seq_len[b] - 1;
    idx = idx < 0 ? 0 : (idx > Sv - 1 ? Sv - 1 : idx);
    const __bf16* last = hfin + (((size_t)idx * Bv + b) * Nn + n) * Uu;

    float logit[NCLS];
#pragma unroll
    for (int c = 0; c < NCLS; ++c) logit[c] = fcb[c];
    for (int u = 0; u < Uu; ++u) {
        float x = (float)last[u];
        x = x > 0.0f ? x : 0.0f;
#pragma unroll
        for (int c = 0; c < NCLS; ++c) logit[c] += x * fcw[u * NCLS + c];
    }
#pragma unroll
    for (int c = 0; c < NCLS; ++c) sm[c * Nn + n] = logit[c];
    __syncthreads();
    for (int st = Nn / 2; st > 0; st >>= 1) {
        if (n < st) {
#pragma unroll
            for (int c = 0; c < NCLS; ++c)
                sm[c * Nn + n] = fmaxf(sm[c * Nn + n], sm[c * Nn + n + st]);
        }
        __syncthreads();
    }
    if (n < NCLS) out[(size_t)b * NCLS + n] = sm[n * Nn];
}

// --------------------------------------------------------------------------
extern "C" void kernel_launch(void* const* d_in, const int* in_sizes, int n_in,
                              void* d_out, int out_size, void* d_ws, size_t ws_size,
                              hipStream_t stream) {
    const float* input_seq = (const float*)d_in[0];
    const int*   seq_len   = (const int*)d_in[1];
    const float* adj       = (const float*)d_in[2];
    const float* Wmsg      = (const float*)d_in[3];   // [L,E,U,U]
    const float* bmsg      = (const float*)d_in[4];   // [L,U]
    const float* Wg        = (const float*)d_in[5];   // [L,3,U,U]
    const float* Ug        = (const float*)d_in[6];   // [L,3,U,U]
    const float* bgp       = (const float*)d_in[7];   // [L,3,U]
    const float* fcw       = (const float*)d_in[8];   // [U,5]
    const float* fcb       = (const float*)d_in[9];   // [5]

    const size_t adjElems = (size_t)Bv * Ee * Nn * Nn;     // 1M
    const size_t hElems   = (size_t)Sv * Bv * Nn * Uu;     // 8M
    __bf16* adjb = (__bf16*)d_ws;                          // ws: ~34 MB bf16
    __bf16* ws0  = adjb + adjElems;
    __bf16* ws1  = ws0 + hElems;

    cvt_adj<<<(unsigned)((adjElems + 255) / 256), 256, 0, stream>>>(adj, adjb);
    ggnn_init<<<(unsigned)((hElems + 255) / 256), 256, 0, stream>>>(input_seq, ws0);

    __bf16* cur = ws0;
    __bf16* nxt = ws1;
    for (int l = 0; l < 2; ++l) {
        const float* Wm_l = Wmsg + (size_t)l * Ee * Uu * Uu;
        const float* bm_l = bmsg + (size_t)l * Uu;
        const float* Wg_l = Wg   + (size_t)l * 3 * Uu * Uu;
        const float* Ug_l = Ug   + (size_t)l * 3 * Uu * Uu;
        const float* bg_l = bgp  + (size_t)l * 3 * Uu;
        for (int st = 0; st < 3; ++st) {
            ggnn_step<<<Sv * Bv, 256, 0, stream>>>(cur, nxt, adjb, Wm_l, bm_l, Wg_l, Ug_l, bg_l);
            __bf16* tmp = cur; cur = nxt; nxt = tmp;
        }
    }
    ggnn_head<<<Bv, 128, 0, stream>>>(cur, seq_len, fcw, fcb, (float*)d_out);
}
